// AttentionLayer_13735305412668
// MI455X (gfx1250) — matrix operand order repaired and checked by reading, not compiled, not run on hardware
//
#include <hip/hip_runtime.h>

#define B_DIM 1024
#define M_DIM 64
#define H_DIM 1024

typedef __attribute__((ext_vector_type(16))) _Float16 v16h;
typedef __attribute__((ext_vector_type(8)))  _Float16 v8h;
typedef __attribute__((ext_vector_type(8)))  float    v8f;

// C[Mrows,N] = X[Mrows,K] * B (+bias).  TRANSB: B = W^T with W[n*K+k] (nn.Linear
// weight layout); !TRANSB: B = W with W[k*N+n].  64x64 C tile per 256-thread block,
// 8 waves, each wave owns a 16x32 strip (two 16x16 WMMA accumulators).
template <bool TRANSB, bool HAS_BIAS>
__global__ __launch_bounds__(256) void gemm_wmma(
    const float* __restrict__ X, const float* __restrict__ W,
    const float* __restrict__ bias, float* __restrict__ C,
    int Mrows, int N, int K) {
  __shared__ _Float16 Alds[64 * 32];   // [m_local][k_local]
  __shared__ _Float16 Blds[64 * 32];   // [n_local][k_local]

  const int t    = threadIdx.x;
  const int lane = t & 31;
  const int wave = t >> 5;
  const int hi   = lane >> 4;   // lane half (selects K sub-block per ISA layout)
  const int ln   = lane & 15;
  const int m0   = blockIdx.y * 64;
  const int n0   = blockIdx.x * 64;
  const int mw   = (wave >> 1) * 16;   // wave's M offset in tile
  const int nw   = (wave & 1) * 32;    // wave's N offset in tile

  v8f acc0 = {};
  v8f acc1 = {};

  const int srow = t >> 2;          // 0..63 staging row
  const int scol = (t & 3) * 8;     // 0,8,16,24 staging col

  for (int k0 = 0; k0 < K; k0 += 32) {
    // ---- stage A tile: 64x32 f32 -> f16, row major
    {
      const float* src = X + (size_t)(m0 + srow) * K + k0 + scol;
      float4 f0 = ((const float4*)src)[0];
      float4 f1 = ((const float4*)src)[1];
      _Float16* dst = &Alds[srow * 32 + scol];
      dst[0] = (_Float16)f0.x; dst[1] = (_Float16)f0.y;
      dst[2] = (_Float16)f0.z; dst[3] = (_Float16)f0.w;
      dst[4] = (_Float16)f1.x; dst[5] = (_Float16)f1.y;
      dst[6] = (_Float16)f1.z; dst[7] = (_Float16)f1.w;
    }
    // ---- stage B tile as [n][k]
    if (TRANSB) {
      const float* src = W + (size_t)(n0 + srow) * K + k0 + scol;
      float4 f0 = ((const float4*)src)[0];
      float4 f1 = ((const float4*)src)[1];
      _Float16* dst = &Blds[srow * 32 + scol];
      dst[0] = (_Float16)f0.x; dst[1] = (_Float16)f0.y;
      dst[2] = (_Float16)f0.z; dst[3] = (_Float16)f0.w;
      dst[4] = (_Float16)f1.x; dst[5] = (_Float16)f1.y;
      dst[6] = (_Float16)f1.z; dst[7] = (_Float16)f1.w;
    } else {
      const int kk  = t >> 3;         // 0..31
      const int nn0 = (t & 7) * 8;    // 0..56
      const float* src = W + (size_t)(k0 + kk) * N + n0 + nn0;
      float4 f0 = ((const float4*)src)[0];
      float4 f1 = ((const float4*)src)[1];
      float vals[8] = {f0.x, f0.y, f0.z, f0.w, f1.x, f1.y, f1.z, f1.w};
#pragma unroll
      for (int j = 0; j < 8; ++j) Blds[(nn0 + j) * 32 + kk] = (_Float16)vals[j];
    }
    __syncthreads();

    // ---- A fragment: lane=M (mod 16); elems 0-7 -> K=8*hi.., elems 8-15 -> K=16+8*hi..
    const _Float16* ap = &Alds[(mw + ln) * 32];
    v8h a0 = *(const v8h*)(ap + 8 * hi);
    v8h a1 = *(const v8h*)(ap + 16 + 8 * hi);
    v16h a;
#pragma unroll
    for (int i = 0; i < 8; ++i) { a[i] = a0[i]; a[i + 8] = a1[i]; }

    // ---- B fragments: lane=N (mod 16); elems 0-7 -> K=8*hi.., elems 8-15 -> K=16+8*hi..
    const _Float16* bp0 = &Blds[(nw + ln) * 32];
    v8h b00 = *(const v8h*)(bp0 + 8 * hi);
    v8h b01 = *(const v8h*)(bp0 + 16 + 8 * hi);
    v16h b0;
#pragma unroll
    for (int i = 0; i < 8; ++i) { b0[i] = b00[i]; b0[i + 8] = b01[i]; }

    const _Float16* bp1 = &Blds[(nw + 16 + ln) * 32];
    v8h b10 = *(const v8h*)(bp1 + 8 * hi);
    v8h b11 = *(const v8h*)(bp1 + 16 + 8 * hi);
    v16h b1;
#pragma unroll
    for (int i = 0; i < 8; ++i) { b1[i] = b10[i]; b1[i + 8] = b11[i]; }

    acc0 = __builtin_amdgcn_wmma_f32_16x16x32_f16(false, a, false, b0,
                                                  (short)0, acc0, false, false);
    acc1 = __builtin_amdgcn_wmma_f32_16x16x32_f16(false, a, false, b1,
                                                  (short)0, acc1, false, false);
    __syncthreads();
  }

  // ---- epilogue: D elem r -> M = r + 8*hi, N = lane%16
#pragma unroll
  for (int r = 0; r < 8; ++r) {
    const int row  = m0 + mw + r + 8 * hi;
    const int col0 = n0 + nw + ln;
    const int col1 = col0 + 16;
    float v0 = acc0[r];
    float v1 = acc1[r];
    if (HAS_BIAS) { v0 += bias[col0]; v1 += bias[col1]; }
    C[(size_t)row * N + col0] = v0;
    C[(size_t)row * N + col1] = v1;
  }
}

// Fused per-batch: qb = Q[b].bk ; scores = (A[b,m,:].Qk[b] + qb)/sqrt(H) * exp(-We*IE);
// softmax over M; write attn; ctxA[b,:] = sum_m attn[m]*A[b,m,:]
__global__ __launch_bounds__(256) void attn_fused(
    const float* __restrict__ A,    // all_states [B,M,H]
    const float* __restrict__ Q,    // [B,H]
    const float* __restrict__ Qk,   // [B,H]
    const float* __restrict__ bk,   // [H]
    const float* __restrict__ IEs,  // [B,M]
    const float* __restrict__ We,   // [M]
    float* __restrict__ attn_out,   // [B,M]
    float* __restrict__ ctxA) {     // [B,H] (may alias Q)
  __shared__ float qks[H_DIM];
  __shared__ float red[256];
  __shared__ float sm[M_DIM];
  __shared__ float aw[M_DIM];

  const int b = blockIdx.x;
  const int t = threadIdx.x;

  float qpart = 0.f;
#pragma unroll
  for (int j = 0; j < 4; ++j) {
    const int h = t + j * 256;
    qks[h] = Qk[(size_t)b * H_DIM + h];
    qpart += Q[(size_t)b * H_DIM + h] * bk[h];
  }
  red[t] = qpart;
  __syncthreads();
  for (int s = 128; s > 0; s >>= 1) {
    if (t < s) red[t] += red[t + s];
    __syncthreads();
  }
  const float qb = red[0];

  // scores: 8 waves x 8 rows
  const int wave = t >> 5, lane = t & 31;
#pragma unroll
  for (int mi = 0; mi < 8; ++mi) {
    const int m = wave * 8 + mi;
    const float* arow = A + ((size_t)b * M_DIM + m) * H_DIM;
    float p = 0.f;
    for (int i = lane; i < H_DIM; i += 32) p += arow[i] * qks[i];
#pragma unroll
    for (int off = 16; off >= 1; off >>= 1) p += __shfl_xor(p, off, 32);
    if (lane == 0) sm[m] = (p + qb) * 0.03125f;  // 1/sqrt(1024)
  }
  __syncthreads();

  if (t < M_DIM) {
    const float v = sm[t] * __expf(-We[t] * IEs[(size_t)b * M_DIM + t]);
    sm[t]  = v;
    red[t] = v;
  }
  __syncthreads();
  for (int s = 32; s > 0; s >>= 1) {
    if (t < s) red[t] = fmaxf(red[t], red[t + s]);
    __syncthreads();
  }
  const float mx = red[0];
  __syncthreads();
  if (t < M_DIM) {
    const float e = __expf(sm[t] - mx);
    aw[t]  = e;
    red[t] = e;
  }
  __syncthreads();
  for (int s = 32; s > 0; s >>= 1) {
    if (t < s) red[t] += red[t + s];
    __syncthreads();
  }
  const float inv = 1.f / red[0];
  __syncthreads();
  if (t < M_DIM) {
    const float av = aw[t] * inv;
    aw[t] = av;
    attn_out[(size_t)b * M_DIM + t] = av;
  }
  __syncthreads();

  float acc[4] = {0.f, 0.f, 0.f, 0.f};
  for (int m = 0; m < M_DIM; ++m) {
    const float am = aw[m];
    const float* arow = A + ((size_t)b * M_DIM + m) * H_DIM;
#pragma unroll
    for (int j = 0; j < 4; ++j) acc[j] += am * arow[t + j * 256];
  }
#pragma unroll
  for (int j = 0; j < 4; ++j) ctxA[(size_t)b * H_DIM + t + j * 256] = acc[j];
}

extern "C" void kernel_launch(void* const* d_in, const int* in_sizes, int n_in,
                              void* d_out, int out_size, void* d_ws, size_t ws_size,
                              hipStream_t stream) {
  const float* last_state = (const float*)d_in[0];
  const float* all_states = (const float*)d_in[1];
  const float* IEs        = (const float*)d_in[2];
  const float* Wq         = (const float*)d_in[3];
  const float* bq         = (const float*)d_in[4];
  const float* Wk         = (const float*)d_in[5];
  const float* bk         = (const float*)d_in[6];
  const float* Wv         = (const float*)d_in[7];
  const float* bv         = (const float*)d_in[8];
  const float* We         = (const float*)d_in[9];

  float* out     = (float*)d_out;
  float* context = out;                               // [B,H]
  float* attn    = out + (size_t)B_DIM * H_DIM;       // [B,M]
  float* Qbuf    = (float*)d_ws;                      // [B,H], later reused as ctxA
  float* Qk      = context;                           // stash Qk where context goes

  dim3 grid(H_DIM / 64, B_DIM / 64), blk(256);

  // 1) Q = last_state @ Wq^T + bq
  gemm_wmma<true, true><<<grid, blk, 0, stream>>>(last_state, Wq, bq, Qbuf,
                                                  B_DIM, H_DIM, H_DIM);
  // 2) Qk = Q @ Wk   (bk folded in as scalar qb inside attn_fused)
  gemm_wmma<false, false><<<grid, blk, 0, stream>>>(Qbuf, Wk, nullptr, Qk,
                                                    B_DIM, H_DIM, H_DIM);
  // 3) scores + softmax + attn + ctxA  (ctxA overwrites Qbuf: per-block reads of
  //    Q[b,:] complete before ctxA[b,:] writes, barrier-separated)
  attn_fused<<<dim3(B_DIM), blk, 0, stream>>>(all_states, Qbuf, Qk, bk, IEs, We,
                                              attn, Qbuf);
  // 4) context = ctxA @ Wv^T + bv  (overwrites the Qk stash, which is now dead)
  gemm_wmma<true, true><<<grid, blk, 0, stream>>>(Qbuf, Wv, bv, context,
                                                  B_DIM, H_DIM, H_DIM);
}
